// BiDAF_16999480557667
// MI455X (gfx1250) — compile-verified, matrix operand order checked by reading, not run on hardware
//
#include <hip/hip_runtime.h>
#include <hip/hip_bf16.h>
#include <stdint.h>

#define Bb 32
#define Tt 2048
#define Jj 256
#define Dd 256

typedef __attribute__((ext_vector_type(16))) __bf16 v16bf;
typedef __attribute__((ext_vector_type(8)))  float  v8f;

union Frag { uint32_t u[8]; v16bf v; };

__device__ __forceinline__ uint32_t pack_bf16(float a, float b) {
  union { float f; uint32_t u; } x, y;
  x.f = a; y.f = b;
  uint32_t ra = (x.u + 0x7FFFu + ((x.u >> 16) & 1u)) >> 16;
  uint32_t rb = (y.u + 0x7FFFu + ((y.u >> 16) & 1u)) >> 16;
  return (rb << 16) | (ra & 0xFFFFu);
}

__device__ __forceinline__ uint16_t f2bf(float a) {
  union { float f; uint32_t u; } x; x.f = a;
  return (uint16_t)((x.u + 0x7FFFu + ((x.u >> 16) & 1u)) >> 16);
}

// ---------------- Kernel 1: bf16 copies of U (row-major + transposed) ----------------
__global__ void prep_u_kernel(const float* __restrict__ U,
                              uint16_t* __restrict__ Ubf,
                              uint16_t* __restrict__ Utbf) {
  int idx = blockIdx.x * blockDim.x + threadIdx.x;   // over B*J*D
  int d = idx % Dd;
  int j = (idx / Dd) % Jj;
  int b = idx / (Dd * Jj);
  uint16_t h = f2bf(U[idx]);
  Ubf[idx] = h;
  Utbf[((size_t)b * Dd + d) * Jj + j] = h;
}

// ---------------- Kernel 2: row dot products (hw = H.w_h, uw = U.w_u) ----------------
__global__ void rowdot_kernel(const float* __restrict__ X,
                              const float* __restrict__ w,
                              float* __restrict__ out, int nrows) {
  int lane = threadIdx.x & 31;
  int warp = threadIdx.x >> 5;
  int row  = blockIdx.x * 8 + warp;
  if (row >= nrows) return;
  const float* xr = X + (size_t)row * Dd;
  float acc = 0.f;
#pragma unroll
  for (int i = 0; i < 8; ++i) acc += xr[lane + 32 * i] * w[lane + 32 * i];
#pragma unroll
  for (int off = 16; off >= 1; off >>= 1) acc += __shfl_down(acc, off, 32);
  if (lane == 0) out[row] = acc;
}

// ---------------- Kernel 3: fused S = Hp @ U^T, masked softmax, c2q = attn @ U -------
__launch_bounds__(64)
__global__ void attn_kernel(const float* __restrict__ H,
                            const uint16_t* __restrict__ Ubf,   // (B,J,D) bf16
                            const uint16_t* __restrict__ Utbf,  // (B,D,J) bf16
                            const float* __restrict__ hw,       // (B,T)
                            const float* __restrict__ uw,       // (B,J)
                            const float* __restrict__ Umask,    // (B,J)
                            const float* __restrict__ w,        // w_hu = w + 2D
                            const float* __restrict__ bias,
                            float* __restrict__ c2q,            // (B,T,D)
                            float* __restrict__ mb)             // (B,T)
{
  __shared__ float sS[2 * 16 * 256];   // 32 KB: 2 waves x (16 rows x 256 f32)
  const int tid  = threadIdx.x;
  const int wave = tid >> 5;
  const int lane = tid & 31;
  const int lo   = lane & 15;
  const int hi   = lane >> 4;
  const int klo  = hi * 8;                 // K sub-offset per ISA A/B bf16 layout
  const int bIdx = blockIdx.x >> 6;        // T/32 = 64 chunks per batch
  const int chunk = blockIdx.x & 63;
  const int rowbase = chunk * 32 + wave * 16;
  float* sRow = sS + wave * 16 * 256;

  const float* whu = w + 2 * Dd;

  // --- A fragments: Hp = H * w_hu, converted f32 -> bf16 on the fly, all 8 k-steps ---
  Frag a[8];
  {
    const float* hrow = H + ((size_t)bIdx * Tt + rowbase + lo) * Dd;
#pragma unroll
    for (int k = 0; k < 8; ++k) {
      const float4* h0 = (const float4*)(hrow + k * 32 + klo);
      const float4* h1 = (const float4*)(hrow + k * 32 + klo + 16);
      const float4* w0 = (const float4*)(whu + k * 32 + klo);
      const float4* w1 = (const float4*)(whu + k * 32 + klo + 16);
      float4 ha = h0[0], hb = h0[1], hc = h1[0], hd = h1[1];
      float4 wa = w0[0], wb = w0[1], wc = w1[0], wd = w1[1];
      a[k].u[0] = pack_bf16(ha.x * wa.x, ha.y * wa.y);
      a[k].u[1] = pack_bf16(ha.z * wa.z, ha.w * wa.w);
      a[k].u[2] = pack_bf16(hb.x * wb.x, hb.y * wb.y);
      a[k].u[3] = pack_bf16(hb.z * wb.z, hb.w * wb.w);
      a[k].u[4] = pack_bf16(hc.x * wc.x, hc.y * wc.y);
      a[k].u[5] = pack_bf16(hc.z * wc.z, hc.w * wc.w);
      a[k].u[6] = pack_bf16(hd.x * wd.x, hd.y * wd.y);
      a[k].u[7] = pack_bf16(hd.z * wd.z, hd.w * wd.w);
    }
  }

  // --- matmul 1: S[16 x 256] = Hp @ U^T, B column n = row n of Ubf (row-major) ---
#pragma unroll 1
  for (int jt = 0; jt < 16; ++jt) {
    v8f acc = {};
    const uint16_t* urow = Ubf + ((size_t)bIdx * Jj + jt * 16 + lo) * Dd;
#pragma unroll
    for (int k = 0; k < 8; ++k) {
      Frag bm;
      uint4 p = *(const uint4*)(urow + k * 32 + klo);
      uint4 q = *(const uint4*)(urow + k * 32 + klo + 16);
      bm.u[0] = p.x; bm.u[1] = p.y; bm.u[2] = p.z; bm.u[3] = p.w;
      bm.u[4] = q.x; bm.u[5] = q.y; bm.u[6] = q.z; bm.u[7] = q.w;
      acc = __builtin_amdgcn_wmma_f32_16x16x32_bf16(
          false, a[k].v, false, bm.v, (short)0, acc, false, false);
    }
#pragma unroll
    for (int v = 0; v < 8; ++v)
      sRow[(v + 8 * hi) * 256 + jt * 16 + lo] = acc[v];
  }
  __syncthreads();

  // --- masked softmax per row (lanes 0..15 each own one full row), also mb row-max ---
  if (lane < 16) {
    const int row = lo;
    const float hwv = hw[(size_t)bIdx * Tt + rowbase + row] + bias[0];
    const float* uwb = uw + bIdx * Jj;
    const float* umb = Umask + bIdx * Jj;
    float* srow = sRow + row * 256;
    float xmax = -3.4e38f, mbmax = -3.4e38f;
    for (int j = 0; j < 256; ++j) {
      float s = srow[j] + hwv + uwb[j];
      float m = umb[j];
      xmax  = fmaxf(xmax, s * m);
      mbmax = fmaxf(mbmax, s * m - 10000000.0f * (1.0f - m));
    }
    float Z = 0.f, Zm = 0.f;
    for (int j = 0; j < 256; ++j) {
      float s = srow[j] + hwv + uwb[j];
      float m = umb[j];
      float e = __expf(s * m - xmax);
      Z += e; Zm += e * m;
    }
    float scale = 1.0f / (Zm + 1e-13f * Z);
    // re-pack attn as bf16 in place (write offset 2j < read offset 4j: safe)
    uint32_t* prow = (uint32_t*)srow;
    for (int j = 0; j < 256; j += 2) {
      float s0 = srow[j]     + hwv + uwb[j];
      float s1 = srow[j + 1] + hwv + uwb[j + 1];
      float m0 = umb[j], m1 = umb[j + 1];
      float p0 = __expf(s0 * m0 - xmax) * m0 * scale;
      float p1 = __expf(s1 * m1 - xmax) * m1 * scale;
      prow[j >> 1] = pack_bf16(p0, p1);
    }
    mb[(size_t)bIdx * Tt + rowbase + row] = mbmax;
  }
  __syncthreads();

  // --- matmul 2: c2q[16 x 256] = attn @ U, A from LDS, B column n = row n of Utbf ---
  Frag a2[8];
  {
    const uint16_t* ar = (const uint16_t*)sRow + (size_t)lo * 512;  // bf16 row stride 512
#pragma unroll
    for (int k = 0; k < 8; ++k) {
      uint4 p = *(const uint4*)(ar + k * 32 + klo);
      uint4 q = *(const uint4*)(ar + k * 32 + klo + 16);
      a2[k].u[0] = p.x; a2[k].u[1] = p.y; a2[k].u[2] = p.z; a2[k].u[3] = p.w;
      a2[k].u[4] = q.x; a2[k].u[5] = q.y; a2[k].u[6] = q.z; a2[k].u[7] = q.w;
    }
  }
#pragma unroll 1
  for (int nt = 0; nt < 16; ++nt) {
    v8f acc = {};
    const uint16_t* utr = Utbf + ((size_t)bIdx * Dd + nt * 16 + lo) * Jj;
#pragma unroll
    for (int k = 0; k < 8; ++k) {
      Frag bm;
      uint4 p = *(const uint4*)(utr + k * 32 + klo);
      uint4 q = *(const uint4*)(utr + k * 32 + klo + 16);
      bm.u[0] = p.x; bm.u[1] = p.y; bm.u[2] = p.z; bm.u[3] = p.w;
      bm.u[4] = q.x; bm.u[5] = q.y; bm.u[6] = q.z; bm.u[7] = q.w;
      acc = __builtin_amdgcn_wmma_f32_16x16x32_bf16(
          false, a2[k].v, false, bm.v, (short)0, acc, false, false);
    }
#pragma unroll
    for (int v = 0; v < 8; ++v)
      c2q[((size_t)bIdx * Tt + rowbase + v + 8 * hi) * Dd + nt * 16 + lo] = acc[v];
  }
}

// ---------------- Kernel 4: b = masked_softmax(mb, H_mask); q2c = b @ H --------------
__global__ void q2c_kernel(const float* __restrict__ H,
                           const float* __restrict__ mb,
                           const float* __restrict__ Hmask,
                           float* __restrict__ q2c) {
  __shared__ float red[256];
  __shared__ float sbt[Tt];
  const int b = blockIdx.x, tid = threadIdx.x;
  const float* mbb = mb + (size_t)b * Tt;
  const float* hm  = Hmask + (size_t)b * Tt;

  float lmax = -3.4e38f;
  for (int i = 0; i < 8; ++i) { int t = tid + i * 256; lmax = fmaxf(lmax, mbb[t] * hm[t]); }
  red[tid] = lmax; __syncthreads();
  for (int s = 128; s >= 1; s >>= 1) {
    if (tid < s) red[tid] = fmaxf(red[tid], red[tid + s]);
    __syncthreads();
  }
  const float xmax = red[0]; __syncthreads();

  float lZ = 0.f, lZm = 0.f;
  for (int i = 0; i < 8; ++i) {
    int t = tid + i * 256;
    float m = hm[t];
    float e = __expf(mbb[t] * m - xmax);
    lZ += e; lZm += e * m;
    sbt[t] = e * m;
  }
  red[tid] = lZ; __syncthreads();
  for (int s = 128; s >= 1; s >>= 1) {
    if (tid < s) red[tid] += red[tid + s];
    __syncthreads();
  }
  const float Z = red[0]; __syncthreads();
  red[tid] = lZm; __syncthreads();
  for (int s = 128; s >= 1; s >>= 1) {
    if (tid < s) red[tid] += red[tid + s];
    __syncthreads();
  }
  const float Zm = red[0]; __syncthreads();

  const float scale = 1.0f / (Zm + 1e-13f * Z);
  for (int i = 0; i < 8; ++i) { int t = tid + i * 256; sbt[t] *= scale; }
  __syncthreads();

  const int d = tid;
  float acc = 0.f;
  const float* hb = H + (size_t)b * Tt * Dd + d;
  for (int t = 0; t < Tt; ++t) acc += sbt[t] * hb[(size_t)t * Dd];
  q2c[(size_t)b * Dd + d] = acc;
}

// ---------------- Kernel 5: G = [H, c2q, H*c2q, H*q2c] (coalesced stream) ------------
__global__ void g_kernel(const float* __restrict__ H,
                         const float* __restrict__ c2q,
                         const float* __restrict__ q2c,
                         float* __restrict__ G) {
  size_t idx = (size_t)blockIdx.x * blockDim.x + threadIdx.x;  // over B*T*D
  int d = (int)(idx % Dd);
  size_t bt = idx / Dd;
  int b = (int)(bt / Tt);
  float h = H[idx];
  float c = c2q[idx];
  float q = q2c[(size_t)b * Dd + d];
  size_t base = bt * (size_t)(4 * Dd);
  G[base + d]          = h;
  G[base + Dd + d]     = c;
  G[base + 2 * Dd + d] = h * c;
  G[base + 3 * Dd + d] = h * q;
}

extern "C" void kernel_launch(void* const* d_in, const int* in_sizes, int n_in,
                              void* d_out, int out_size, void* d_ws, size_t ws_size,
                              hipStream_t stream) {
  const float* U     = (const float*)d_in[0];
  const float* H     = (const float*)d_in[1];
  const float* Umask = (const float*)d_in[2];
  const float* Hmask = (const float*)d_in[3];
  const float* w     = (const float*)d_in[4];
  const float* bias  = (const float*)d_in[5];

  float* G   = (float*)d_out;                       // (B,T,4D)
  float* c2q = G + (size_t)Bb * Tt * 4 * Dd;        // (B,T,D)
  float* q2c = c2q + (size_t)Bb * Tt * Dd;          // (B,D)

  char* ws = (char*)d_ws;
  uint16_t* Ubf  = (uint16_t*)ws;                         // 4 MB (B,J,D) bf16
  uint16_t* Utbf = (uint16_t*)(ws + ((size_t)4 << 20));   // 4 MB (B,D,J) bf16
  float* hw   = (float*)(ws + ((size_t)8 << 20));         // (B,T)
  float* uw   = hw + (size_t)Bb * Tt;                     // (B,J)
  float* mbuf = uw + (size_t)Bb * Jj;                     // (B,T)

  prep_u_kernel<<<(Bb * Jj * Dd) / 256, 256, 0, stream>>>(U, Ubf, Utbf);
  rowdot_kernel<<<(Bb * Tt) / 8, 256, 0, stream>>>(H, w, hw, Bb * Tt);
  rowdot_kernel<<<(Bb * Jj) / 8, 256, 0, stream>>>(U, w + Dd, uw, Bb * Jj);
  attn_kernel<<<Bb * (Tt / 32), 64, 0, stream>>>(H, Ubf, Utbf, hw, uw, Umask, w, bias,
                                                 c2q, mbuf);
  q2c_kernel<<<Bb, 256, 0, stream>>>(H, mbuf, Hmask, q2c);
  g_kernel<<<(Bb * Tt * Dd) / 256, 256, 0, stream>>>(H, c2q, q2c, G);
}
